// DeepEMD_14783277433104
// MI455X (gfx1250) — compile-verified, matrix operand order checked by reading, not run on hardware
//
#include <hip/hip_runtime.h>
#include <hip/hip_bf16.h>
#include <math.h>

typedef __attribute__((ext_vector_type(16))) _Float16 v16h;
typedef __attribute__((ext_vector_type(8)))  _Float16 v8h;
typedef __attribute__((ext_vector_type(8)))  float    v8f;

#define WMMA_F16(a, b, c) \
  __builtin_amdgcn_wmma_f32_16x16x32_f16(false, (a), false, (b), (short)0, (c), false, false)

// ---------------------------------------------------------------------------
// WMMA GEMM core: C[M,N] (f32) = A[M,K] x B[K,N] (+bias[N]).
// Operands are pre-converted f16 in global memory ([row, K] contiguous, K%8==0)
// via loaders exposing load8(row,k) -> v8h (one global_load_b128each).
// Block: 256 threads (8 waves), tile 128x64, K-step 32.
// Wave tile 32x32: 2 A frags x 2 B frags -> 4 wmma / k-step / wave.
// ---------------------------------------------------------------------------
__device__ __forceinline__ v16h combine16(v8h lo, v8h hi) {
  v16h r;
  #pragma unroll
  for (int i = 0; i < 8; ++i) { r[i] = lo[i]; r[i + 8] = hi[i]; }
  return r;
}

struct AH { const _Float16* A; int ld;      // [M,K] f16 row-major
  __device__ v8h load8(int m, int k) const { return *(const v8h*)(A + (size_t)m * ld + k); }
  __device__ void pf(int m, int k) const { __builtin_prefetch(A + (size_t)m * ld + k, 0, 0); } };

struct BH { const _Float16* B; int ld;      // [N,K] f16 row-major
  __device__ v8h load8(int n, int k) const { return *(const v8h*)(B + (size_t)n * ld + k); }
  __device__ void pf(int n, int k) const { __builtin_prefetch(B + (size_t)n * ld + k, 0, 0); } };

// conv1 implicit im2col, K-order (kh,kw,ci), ci in [0,1280): source NHWC f16
// ZQ[375][5][5][1280] (fused support+noise | query). Zero padding by mask.
struct AC1H { const _Float16* zq;
  __device__ v8h load8(int m, int k) const {
    int img = m / 25, pix = m % 25, oh = pix / 5, ow = pix % 5;
    int t = k / 1280, ci = k % 1280, kh = t / 3, kw = t % 3;
    int h = oh + kh - 1, w = ow + kw - 1;
    bool inb = ((unsigned)h < 5u) && ((unsigned)w < 5u);
    int p = inb ? (h * 5 + w) : 0;
    v8h v = *(const v8h*)(zq + ((size_t)img * 25 + p) * 1280 + ci);
    return v * (inb ? (_Float16)1.f : (_Float16)0.f);
  }
  __device__ void pf(int, int) const {} };

// conv2 implicit im2col, K-order (kh,kw,ci), ci in [0,320): source NHWC f16
// h1[375][2][2][320]. Zero padding by mask.
struct AC2H { const _Float16* h1;
  __device__ v8h load8(int m, int k) const {
    int img = m / 4, pix = m % 4, oh = pix / 2, ow = pix % 2;
    int t = k / 320, ci = k % 320, kh = t / 3, kw = t % 3;
    int h = oh + kh - 1, w = ow + kw - 1;
    bool inb = ((unsigned)h < 2u) && ((unsigned)w < 2u);
    int hc = inb ? h : 0, wc = inb ? w : 0;
    v8h v = *(const v8h*)(h1 + (((size_t)img * 2 + hc) * 2 + wc) * 320 + ci);
    return v * (inb ? (_Float16)1.f : (_Float16)0.f);
  }
  __device__ void pf(int, int) const {} };

template <typename AL, typename BL>
__global__ __launch_bounds__(256) void gemm_wmma(AL aload, BL bload,
    float* __restrict__ C, const float* __restrict__ bias, int M, int N, int K)
{
  __shared__ _Float16 sA[128][32];   // [m][k]
  __shared__ _Float16 sB[64][32];    // [n][k]
  const int tid = threadIdx.x;
  const int lane = tid & 31, wave = tid >> 5;
  const int wm = wave & 3, wn = wave >> 2;          // wave -> 32-row, 32-col tile
  const int m0 = blockIdx.y * 128, n0 = blockIdx.x * 64;
  const int lm = lane & 15;
  const int kb = (lane >> 4) << 3;                  // 0 or 8 (wave32 frag layout)

  // A staging: thread -> row tid>>1 (0..127), 16 halfs at (tid&1)*16
  const int ar = tid >> 1, ak = (tid & 1) << 4;
  const int gmA = m0 + ar;
  const _Float16 amask = (gmA < M) ? (_Float16)1.f : (_Float16)0.f;
  const int gmC = (gmA < M) ? gmA : (M - 1);
  // B staging: thread -> row tid>>2 (0..63), 8 halfs at (tid&3)*8
  const int br = tid >> 2, bk = (tid & 3) << 3;
  const int gnB = n0 + br;
  const _Float16 bmask = (gnB < N) ? (_Float16)1.f : (_Float16)0.f;
  const int gnC = (gnB < N) ? gnB : (N - 1);

  v8f acc00 = {}, acc01 = {}, acc10 = {}, acc11 = {};
  for (int k0 = 0; k0 < K; k0 += 32) {
    v8h a0 = aload.load8(gmC, k0 + ak) * amask;
    v8h a1 = aload.load8(gmC, k0 + ak + 8) * amask;
    v8h b0 = bload.load8(gnC, k0 + bk) * bmask;
    if (k0 + 32 < K) { aload.pf(gmC, k0 + 32 + ak); bload.pf(gnC, k0 + 32 + bk); }
    __syncthreads();                                // prior fragment reads done
    *(v8h*)&sA[ar][ak]     = a0;
    *(v8h*)&sA[ar][ak + 8] = a1;
    *(v8h*)&sB[br][bk]     = b0;
    __syncthreads();

    v16h af0, af1, bf0, bf1;
    { const _Float16* p = &sA[wm * 32 + lm][0];      af0 = combine16(*(const v8h*)(p + kb), *(const v8h*)(p + kb + 16)); }
    { const _Float16* p = &sA[wm * 32 + 16 + lm][0]; af1 = combine16(*(const v8h*)(p + kb), *(const v8h*)(p + kb + 16)); }
    { const _Float16* p = &sB[wn * 32 + lm][0];      bf0 = combine16(*(const v8h*)(p + kb), *(const v8h*)(p + kb + 16)); }
    { const _Float16* p = &sB[wn * 32 + 16 + lm][0]; bf1 = combine16(*(const v8h*)(p + kb), *(const v8h*)(p + kb + 16)); }
    acc00 = WMMA_F16(af0, bf0, acc00);
    acc01 = WMMA_F16(af0, bf1, acc01);
    acc10 = WMMA_F16(af1, bf0, acc10);
    acc11 = WMMA_F16(af1, bf1, acc11);
  }

  #pragma unroll
  for (int ms = 0; ms < 2; ++ms) {
    #pragma unroll
    for (int nh = 0; nh < 2; ++nh) {
      v8f& acc = ms ? (nh ? acc11 : acc10) : (nh ? acc01 : acc00);
      int n = n0 + wn * 32 + nh * 16 + lm;
      if (n < N) {
        float bv = bias ? bias[n] : 0.f;
        #pragma unroll
        for (int r = 0; r < 8; ++r) {
          int m = m0 + wm * 32 + ms * 16 + r + ((lane >> 4) << 3);
          if (m < M) C[(size_t)m * N + n] = acc[r] + bv;
        }
      }
    }
  }
}

// ---------------------------------------------------------------------------
// Pre-processing / gather / elementwise kernels
// ---------------------------------------------------------------------------
__global__ void cvt_f16_kernel(const float* __restrict__ in, _Float16* __restrict__ out, int n8) {
  int i = blockIdx.x * blockDim.x + threadIdx.x;   // each handles 8 contiguous
  if (i >= n8) return;
  const float* p = in + (size_t)i * 8;
  float4 x = *(const float4*)p, y = *(const float4*)(p + 4);
  v8h o = { (_Float16)x.x, (_Float16)x.y, (_Float16)x.z, (_Float16)x.w,
            (_Float16)y.x, (_Float16)y.y, (_Float16)y.z, (_Float16)y.w };
  *(v8h*)(out + (size_t)i * 8) = o;
}

__global__ void transpose640(const float* __restrict__ W, _Float16* __restrict__ WT) {
  int i = blockIdx.x * blockDim.x + threadIdx.x;          // out [N=640][K=640] f16
  if (i >= 640 * 640) return;
  int n = i / 640, k = i % 640;
  WT[i] = (_Float16)W[(size_t)k * 640 + n];
}

// fused conv1 input, NHWC f16: ZQ[img=(b,s)][p][ci] ; ci<640: support+noise[s], else query[b]
__global__ void build_zq(const float* __restrict__ sup, const float* __restrict__ noi,
                         const float* __restrict__ qry, _Float16* __restrict__ zq) {
  int i = blockIdx.x * blockDim.x + threadIdx.x;          // 375*25*1280
  if (i >= 375 * 25 * 1280) return;
  int ci = i % 1280, p = (i / 1280) % 25, img = i / 32000;
  int b = img / 5, s = img % 5;
  float v = (ci < 640) ? (sup[((size_t)s * 640 + ci) * 25 + p] + noi[((size_t)s * 640 + ci) * 25 + p])
                       : qry[((size_t)b * 640 + (ci - 640)) * 25 + p];
  zq[i] = (_Float16)v;
}

// c1_w [320][1280][3][3] -> wr1 f16 [320][kh*3+kw][1280]
__global__ void reorder_c1(const float* __restrict__ W, _Float16* __restrict__ R) {
  int i = blockIdx.x * blockDim.x + threadIdx.x;
  if (i >= 320 * 11520) return;
  int o = i / 11520, r = i % 11520, t = r / 1280, ci = r % 1280, kh = t / 3, kw = t % 3;
  R[i] = (_Float16)W[(((size_t)o * 1280 + ci) * 3 + kh) * 3 + kw];
}

// c2_w [64][320][3][3] -> wr2 f16 [64][kh*3+kw][320]
__global__ void reorder_c2(const float* __restrict__ W, _Float16* __restrict__ R) {
  int i = blockIdx.x * blockDim.x + threadIdx.x;
  if (i >= 64 * 2880) return;
  int o = i / 2880, r = i % 2880, t = r / 320, ci = r % 320, kh = t / 3, kw = t % 3;
  R[i] = (_Float16)W[(((size_t)o * 320 + ci) * 3 + kh) * 3 + kw];
}

__global__ void build_xs(const float* __restrict__ sup, _Float16* __restrict__ Xs) {
  int i = blockIdx.x * blockDim.x + threadIdx.x;          // [125,640] f16
  if (i >= 125 * 640) return;
  int c = i % 640, row = i / 640, b = row / 25, n = row % 25;
  Xs[i] = (_Float16)sup[((size_t)b * 640 + c) * 25 + n];
}

__global__ void blend_support(const float* __restrict__ s0, const float* __restrict__ satt,
                              float* __restrict__ so) {
  int i = blockIdx.x * blockDim.x + threadIdx.x;          // [5,640,25]
  if (i >= 5 * 640 * 25) return;
  int n = i % 25, c = (i / 25) % 640, b = i / 16000;
  so[i] = 0.5f * s0[i] + 0.5f * satt[((size_t)b * 25 + n) * 640 + c];
}

__global__ void build_xa(const float* __restrict__ qry, const float* __restrict__ sorig,
                         _Float16* __restrict__ XA) {
  int i = blockIdx.x * blockDim.x + threadIdx.x;          // [1875,1280] f16
  if (i >= 1875 * 1280) return;
  int c = i % 1280, row = i / 1280, b = row / 25, n = row % 25;
  float v = (c < 640) ? qry[((size_t)b * 640 + c) * 25 + n]
                      : sorig[((size_t)(b % 5) * 640 + (c - 640)) * 25 + n];
  XA[i] = (_Float16)v;
}

__global__ void blend_query(const float* __restrict__ qry, const float* __restrict__ qatt,
                            float* __restrict__ qo) {
  int i = blockIdx.x * blockDim.x + threadIdx.x;          // [75,640,25]
  if (i >= 75 * 640 * 25) return;
  int n = i % 25, c = (i / 25) % 640, b = i / 16000;
  qo[i] = 0.5f * qry[i] + 0.5f * qatt[((size_t)b * 25 + n) * 640 + c];
}

// single-head attention on [nb,25,640] (softmax(QK^T/sqrt(640)) V), one block/batch
__global__ __launch_bounds__(256) void attention_kernel(
    const float* __restrict__ Q, const float* __restrict__ Kp,
    const float* __restrict__ V, float* __restrict__ O)
{
  __shared__ float sc[625];
  int b = blockIdx.x;
  const float scale = rsqrtf(640.f);
  const float* q = Q + (size_t)b * 25 * 640;
  const float* k = Kp + (size_t)b * 25 * 640;
  for (int p = threadIdx.x; p < 625; p += blockDim.x) {
    int n = p / 25, m = p % 25;
    const float4* qr = (const float4*)(q + n * 640);
    const float4* kr = (const float4*)(k + m * 640);
    float s = 0.f;
    for (int c = 0; c < 160; ++c) {
      float4 a = qr[c], d = kr[c];
      s += a.x * d.x + a.y * d.y + a.z * d.z + a.w * d.w;
    }
    sc[p] = s * scale;
  }
  __syncthreads();
  if (threadIdx.x < 25) {
    int n = threadIdx.x;
    float mx = -1e30f;
    for (int m = 0; m < 25; ++m) mx = fmaxf(mx, sc[n * 25 + m]);
    float sm = 0.f;
    for (int m = 0; m < 25; ++m) { float e = expf(sc[n * 25 + m] - mx); sc[n * 25 + m] = e; sm += e; }
    float inv = 1.f / sm;
    for (int m = 0; m < 25; ++m) sc[n * 25 + m] *= inv;
  }
  __syncthreads();
  const float* v = V + (size_t)b * 25 * 640;
  float* o = O + (size_t)b * 25 * 640;
  for (int p = threadIdx.x; p < 25 * 640; p += blockDim.x) {
    int n = p / 640, c = p % 640;
    float s = 0.f;
    for (int m = 0; m < 25; ++m) s += sc[n * 25 + m] * v[m * 640 + c];
    o[p] = s;
  }
}

// batch-norm statistics over rows (layout [rows, C]); one block per channel
__global__ __launch_bounds__(256) void bn_stats(const float* __restrict__ Y, int rows, int C,
                                                float* __restrict__ mean, float* __restrict__ var) {
  int ch = blockIdx.x;
  __shared__ float rs[256], rq[256];
  float s = 0.f, s2 = 0.f;
  for (int r = threadIdx.x; r < rows; r += blockDim.x) {
    float v = Y[(size_t)r * C + ch]; s += v; s2 += v * v;
  }
  rs[threadIdx.x] = s; rq[threadIdx.x] = s2; __syncthreads();
  for (int o = 128; o > 0; o >>= 1) {
    if (threadIdx.x < o) { rs[threadIdx.x] += rs[threadIdx.x + o]; rq[threadIdx.x] += rq[threadIdx.x + o]; }
    __syncthreads();
  }
  if (threadIdx.x == 0) {
    float m = rs[0] / rows; mean[ch] = m; var[ch] = rq[0] / rows - m * m;
  }
}

// BN + ReLU + 2x2 maxpool on conv1 output [9375,320]; h1 NHWC f16 [375][2][2][320]
__global__ void bnpool1(const float* __restrict__ Y, const float* __restrict__ mean,
                        const float* __restrict__ var, const float* __restrict__ g,
                        const float* __restrict__ bb, _Float16* __restrict__ h1) {
  int i = blockIdx.x * blockDim.x + threadIdx.x;
  if (i >= 375 * 2 * 2 * 320) return;
  int ch = i % 320, x = (i / 320) % 2, y = (i / 640) % 2, img = i / 1280;
  float inv = rsqrtf(var[ch] + 1e-5f), gm = g[ch], bt = bb[ch], mn = mean[ch];
  float best = 0.f;
  #pragma unroll
  for (int dy = 0; dy < 2; ++dy)
    #pragma unroll
    for (int dx = 0; dx < 2; ++dx) {
      int pix = (2 * y + dy) * 5 + (2 * x + dx);
      float v = (Y[((size_t)img * 25 + pix) * 320 + ch] - mn) * inv * gm + bt;
      best = fmaxf(best, fmaxf(v, 0.f));
    }
  h1[i] = (_Float16)best;
}

// BN + ReLU + pool on conv2 output [1500,64] -> h2 [375,64]
__global__ void bnpool2(const float* __restrict__ Y, const float* __restrict__ mean,
                        const float* __restrict__ var, const float* __restrict__ g,
                        const float* __restrict__ bb, float* __restrict__ h2) {
  int i = blockIdx.x * blockDim.x + threadIdx.x;
  if (i >= 375 * 64) return;
  int ch = i % 64, img = i / 64;
  float inv = rsqrtf(var[ch] + 1e-5f), gm = g[ch], bt = bb[ch], mn = mean[ch];
  float best = 0.f;
  #pragma unroll
  for (int pix = 0; pix < 4; ++pix) {
    float v = (Y[((size_t)img * 4 + pix) * 64 + ch] - mn) * inv * gm + bt;
    best = fmaxf(best, fmaxf(v, 0.f));
  }
  h2[i] = best;
}

__global__ void fc_kernel(const float* __restrict__ h2, const float* __restrict__ w1,
                          const float* __restrict__ b1, const float* __restrict__ w2,
                          const float* __restrict__ b2, float* __restrict__ score) {
  int img = blockIdx.x * blockDim.x + threadIdx.x;
  if (img >= 375) return;
  const float* h = h2 + (size_t)img * 64;
  float out = b2[0];
  #pragma unroll
  for (int j = 0; j < 8; ++j) {
    float a = b1[j];
    for (int c = 0; c < 64; ++c) a += h[c] * w1[j * 64 + c];
    out += fmaxf(a, 0.f) * w2[j];
  }
  score[img] = out;
}

// mean over last dim of 25 ([B,640,25] -> [B,640])
__global__ void pool_mean(const float* __restrict__ X, float* __restrict__ O, int total) {
  int i = blockIdx.x * blockDim.x + threadIdx.x;
  if (i >= total) return;
  const float* p = X + (size_t)i * 25;
  float s = 0.f;
  for (int n = 0; n < 25; ++n) s += p[n];
  O[i] = s * (1.f / 25.f);
}

__global__ void compute_w1(const float* __restrict__ qf, const float* __restrict__ pp,
                           float* __restrict__ w1) {
  int i = blockIdx.x * blockDim.x + threadIdx.x;   // [75,5,25]
  if (i >= 9375) return;
  int p = i % 25, s = (i / 25) % 5, b = i / 125;
  float acc = 0.f;
  for (int c = 0; c < 640; ++c) acc += qf[((size_t)b * 640 + c) * 25 + p] * pp[s * 640 + c];
  w1[i] = fmaxf(acc, 0.f) + 0.001f;
}

__global__ void compute_w2(const float* __restrict__ pf, const float* __restrict__ qp,
                           float* __restrict__ w2) {
  int i = blockIdx.x * blockDim.x + threadIdx.x;   // [5,75,25]
  if (i >= 9375) return;
  int p = i % 25, b = (i / 25) % 75, s = i / 1875;
  float acc = 0.f;
  for (int c = 0; c < 640; ++c) acc += pf[((size_t)s * 640 + c) * 25 + p] * qp[(size_t)b * 640 + c];
  w2[i] = fmaxf(acc, 0.f) + 0.001f;
}

// channel-center + L2-normalize each spatial column; QN f16 row-major [(b,p),c]
__global__ __launch_bounds__(128) void normalize_q_kernel(const float* __restrict__ qf,
                                                          _Float16* __restrict__ QN) {
  int bp = blockIdx.x;                       // 0..1874
  int b = bp / 25, p = bp % 25;
  const size_t base = ((size_t)b * 640) * 25 + p;
  __shared__ float red[128];
  int t = threadIdx.x;
  float s = 0.f;
  for (int c = t; c < 640; c += 128) s += qf[base + (size_t)c * 25];
  red[t] = s; __syncthreads();
  for (int o = 64; o > 0; o >>= 1) { if (t < o) red[t] += red[t + o]; __syncthreads(); }
  float mean = red[0] * (1.f / 640.f); __syncthreads();
  float s2 = 0.f;
  for (int c = t; c < 640; c += 128) { float v = qf[base + (size_t)c * 25] - mean; s2 += v * v; }
  red[t] = s2; __syncthreads();
  for (int o = 64; o > 0; o >>= 1) { if (t < o) red[t] += red[t + o]; __syncthreads(); }
  float inv = 1.f / fmaxf(sqrtf(red[0]), 1e-8f);
  for (int c = t; c < 640; c += 128) QN[(size_t)bp * 640 + c] = (_Float16)((qf[base + (size_t)c * 25] - mean) * inv);
}

// same for prototypes; PN f16 row-major [(s,p),c] -> BH ([N,K]) operand
__global__ __launch_bounds__(128) void normalize_p_kernel(const float* __restrict__ pf,
                                                          _Float16* __restrict__ PN) {
  int sp = blockIdx.x;                       // 0..124
  int s = sp / 25, p = sp % 25;
  const size_t base = ((size_t)s * 640) * 25 + p;
  __shared__ float red[128];
  int t = threadIdx.x;
  float sm = 0.f;
  for (int c = t; c < 640; c += 128) sm += pf[base + (size_t)c * 25];
  red[t] = sm; __syncthreads();
  for (int o = 64; o > 0; o >>= 1) { if (t < o) red[t] += red[t + o]; __syncthreads(); }
  float mean = red[0] * (1.f / 640.f); __syncthreads();
  float s2 = 0.f;
  for (int c = t; c < 640; c += 128) { float v = pf[base + (size_t)c * 25] - mean; s2 += v * v; }
  red[t] = s2; __syncthreads();
  for (int o = 64; o > 0; o >>= 1) { if (t < o) red[t] += red[t + o]; __syncthreads(); }
  float inv = 1.f / fmaxf(sqrtf(red[0]), 1e-8f);
  for (int c = t; c < 640; c += 128) PN[(size_t)sp * 640 + c] = (_Float16)((pf[base + (size_t)c * 25] - mean) * inv);
}

__global__ void marginals(const float* __restrict__ w1, const float* __restrict__ w2,
                          float* __restrict__ rm, float* __restrict__ cm) {
  int i = blockIdx.x * blockDim.x + threadIdx.x;   // (b,s) pair
  if (i >= 375) return;
  int b = i / 5, s = i % 5;
  float sr = 0.f;
  for (int p = 0; p < 25; ++p) sr += w1[(size_t)i * 25 + p];
  float fr = 25.f / sr;
  for (int p = 0; p < 25; ++p) rm[(size_t)i * 25 + p] = w1[(size_t)i * 25 + p] * fr;
  const float* wp = w2 + ((size_t)s * 75 + b) * 25;
  float sc = 0.f;
  for (int p = 0; p < 25; ++p) sc += wp[p];
  float fc = 25.f / sc;
  for (int p = 0; p < 25; ++p) cm[(size_t)i * 25 + p] = wp[p] * fc;
}

// log-domain Sinkhorn, 50 iters, eps=0.1; one wave per (b,s); writes final outputs
__global__ __launch_bounds__(32) void sinkhorn_kernel(const float* __restrict__ sim2,
    const float* __restrict__ rm, const float* __restrict__ cm,
    const float* __restrict__ score, float* __restrict__ out) {
  int bs = blockIdx.x;                 // 0..374, bs = b*5 + s
  int b = bs / 5, s = bs % 5;
  __shared__ float cost[625], f[25], g[25], lr[25], lc[25];
  int t = threadIdx.x;
  for (int i = t; i < 625; i += 32) {
    int q = i / 25, p = i % 25;
    cost[i] = 1.f - sim2[((size_t)b * 25 + q) * 125 + s * 25 + p];
  }
  if (t < 25) { lr[t] = logf(rm[(size_t)bs * 25 + t]); lc[t] = logf(cm[(size_t)bs * 25 + t]);
                f[t] = 0.f; g[t] = 0.f; }
  __syncthreads();
  const float eps = 0.1f, inveps = 10.f;
  for (int it = 0; it < 50; ++it) {
    if (t < 25) {
      float mx = -1e30f; float vals[25];
      #pragma unroll
      for (int p = 0; p < 25; ++p) { float v = (g[p] - cost[t * 25 + p]) * inveps; vals[p] = v; mx = fmaxf(mx, v); }
      float sm = 0.f;
      #pragma unroll
      for (int p = 0; p < 25; ++p) sm += expf(vals[p] - mx);
      f[t] = eps * lr[t] - eps * (mx + logf(sm));
    }
    __syncthreads();
    if (t < 25) {
      float mx = -1e30f; float vals[25];
      #pragma unroll
      for (int q = 0; q < 25; ++q) { float v = (f[q] - cost[q * 25 + t]) * inveps; vals[q] = v; mx = fmaxf(mx, v); }
      float sm = 0.f;
      #pragma unroll
      for (int q = 0; q < 25; ++q) sm += expf(vals[q] - mx);
      g[t] = eps * lc[t] - eps * (mx + logf(sm));
    }
    __syncthreads();
  }
  float acc = 0.f;
  for (int i = t; i < 625; i += 32) {
    int q = i / 25, p = i % 25;
    float c0 = cost[i];
    acc += (1.f - c0) * expf((f[q] + g[p] - c0) * inveps);
  }
  for (int o = 16; o > 0; o >>= 1) acc += __shfl_down(acc, o);
  if (t == 0) {
    float sc = score[bs];
    out[bs]       = 0.75f * acc * 0.5f + 0.25f * sc * 12.5f;  // (1-LAMDA)*emd*(TEMP/NN)+LAMDA*score*TEMP2
    out[375 + bs] = sc / 12.5f;                                // score / TAU
  }
}

// ---------------------------------------------------------------------------
static inline float* wsAlloc(char*& p, size_t nfloats) {
  float* r = (float*)p;
  p += ((nfloats * sizeof(float) + 255) & ~(size_t)255);
  return r;
}
static inline _Float16* wsAllocH(char*& p, size_t nhalfs) {
  _Float16* r = (_Float16*)p;
  p += ((nhalfs * sizeof(_Float16) + 255) & ~(size_t)255);
  return r;
}

extern "C" void kernel_launch(void* const* d_in, const int* in_sizes, int n_in,
                              void* d_out, int out_size, void* d_ws, size_t ws_size,
                              hipStream_t stream) {
  const float* support = (const float*)d_in[0];   // [1,5,640,5,5]
  const float* query   = (const float*)d_in[1];   // [75,640,5,5]
  const float* noise   = (const float*)d_in[2];   // [1,5,640,5,5]
  const float* wq      = (const float*)d_in[3];
  const float* wk      = (const float*)d_in[4];
  const float* wv      = (const float*)d_in[5];
  const float* wo      = (const float*)d_in[6];
  const float* conv_w  = (const float*)d_in[7];   // [640,1280] = [N,K]
  const float* conv_b  = (const float*)d_in[8];
  const float* c1_w    = (const float*)d_in[9];   // [320,1280,3,3]
  const float* c1_b    = (const float*)d_in[10];
  const float* bn1_g   = (const float*)d_in[11];
  const float* bn1_b   = (const float*)d_in[12];
  const float* c2_w    = (const float*)d_in[13];  // [64,320,3,3]
  const float* c2_b    = (const float*)d_in[14];
  const float* bn2_g   = (const float*)d_in[15];
  const float* bn2_b   = (const float*)d_in[16];
  const float* fc1_w   = (const float*)d_in[17];  // [8,64]
  const float* fc1_b   = (const float*)d_in[18];
  const float* fc2_w   = (const float*)d_in[19];  // [1,8]
  const float* fc2_b   = (const float*)d_in[20];
  float* out = (float*)d_out;

  char* wp = (char*)d_ws;
  _Float16* wqTH   = wsAllocH(wp, 640 * 640);
  _Float16* wkTH   = wsAllocH(wp, 640 * 640);
  _Float16* wvTH   = wsAllocH(wp, 640 * 640);
  _Float16* woTH   = wsAllocH(wp, 640 * 640);
  _Float16* convWH = wsAllocH(wp, (size_t)640 * 1280);
  _Float16* ZQ     = wsAllocH(wp, (size_t)375 * 25 * 1280);
  _Float16* wr1    = wsAllocH(wp, (size_t)320 * 11520);
  _Float16* wr2    = wsAllocH(wp, (size_t)64 * 2880);
  _Float16* XsH    = wsAllocH(wp, 125 * 640);
  _Float16* attsH  = wsAllocH(wp, 125 * 640);
  _Float16* XAH    = wsAllocH(wp, (size_t)1875 * 1280);
  _Float16* cqcH   = wsAllocH(wp, (size_t)1875 * 640);
  _Float16* attqH  = wsAllocH(wp, (size_t)1875 * 640);
  _Float16* h1H    = wsAllocH(wp, (size_t)375 * 4 * 320);
  _Float16* QNH    = wsAllocH(wp, (size_t)1875 * 640);
  _Float16* PNH    = wsAllocH(wp, (size_t)125 * 640);
  float* Qs    = wsAlloc(wp, 125 * 640);
  float* Ks    = wsAlloc(wp, 125 * 640);
  float* Vs    = wsAlloc(wp, 125 * 640);
  float* atts  = wsAlloc(wp, 125 * 640);
  float* satt  = wsAlloc(wp, 125 * 640);
  float* sorig = wsAlloc(wp, 5 * 640 * 25);
  float* cqc   = wsAlloc(wp, (size_t)1875 * 640);
  float* Qq    = wsAlloc(wp, (size_t)1875 * 640);
  float* Kq    = wsAlloc(wp, (size_t)1875 * 640);
  float* Vq    = wsAlloc(wp, (size_t)1875 * 640);
  float* attq  = wsAlloc(wp, (size_t)1875 * 640);
  float* qatt  = wsAlloc(wp, (size_t)1875 * 640);
  float* qorig = wsAlloc(wp, (size_t)75 * 640 * 25);
  float* Y1    = wsAlloc(wp, (size_t)9375 * 320);
  float* mean1 = wsAlloc(wp, 320);
  float* var1  = wsAlloc(wp, 320);
  float* Y2    = wsAlloc(wp, (size_t)1500 * 64);
  float* mean2 = wsAlloc(wp, 64);
  float* var2  = wsAlloc(wp, 64);
  float* h2    = wsAlloc(wp, (size_t)375 * 64);
  float* score = wsAlloc(wp, 375);
  float* pp    = wsAlloc(wp, 5 * 640);
  float* qp    = wsAlloc(wp, 75 * 640);
  float* w1b   = wsAlloc(wp, 9375);
  float* w2b   = wsAlloc(wp, 9375);
  float* sim2  = wsAlloc(wp, (size_t)1875 * 125);
  float* rm    = wsAlloc(wp, 9375);
  float* cm    = wsAlloc(wp, 9375);

  const dim3 B256(256);
  auto blocks = [](size_t n) { return dim3((unsigned)((n + 255) / 256)); };
  auto ggrid  = [](int M, int N) { return dim3((N + 63) / 64, (M + 127) / 128); };

  // --- pre-processing: f16 weight/activation panels ---
  transpose640<<<blocks(640 * 640), B256, 0, stream>>>(wq, wqTH);
  transpose640<<<blocks(640 * 640), B256, 0, stream>>>(wk, wkTH);
  transpose640<<<blocks(640 * 640), B256, 0, stream>>>(wv, wvTH);
  transpose640<<<blocks(640 * 640), B256, 0, stream>>>(wo, woTH);
  cvt_f16_kernel<<<blocks(640 * 1280 / 8), B256, 0, stream>>>(conv_w, convWH, 640 * 1280 / 8);
  build_zq<<<blocks((size_t)375 * 25 * 1280), B256, 0, stream>>>(support, noise, query, ZQ);
  reorder_c1<<<blocks((size_t)320 * 11520), B256, 0, stream>>>(c1_w, wr1);
  reorder_c2<<<blocks((size_t)64 * 2880), B256, 0, stream>>>(c2_w, wr2);

  // --- support agent attention ---
  build_xs<<<blocks(125 * 640), B256, 0, stream>>>(support, XsH);
  gemm_wmma<AH, BH><<<ggrid(125, 640), B256, 0, stream>>>(AH{XsH, 640}, BH{wqTH, 640}, Qs, nullptr, 125, 640, 640);
  gemm_wmma<AH, BH><<<ggrid(125, 640), B256, 0, stream>>>(AH{XsH, 640}, BH{wkTH, 640}, Ks, nullptr, 125, 640, 640);
  gemm_wmma<AH, BH><<<ggrid(125, 640), B256, 0, stream>>>(AH{XsH, 640}, BH{wvTH, 640}, Vs, nullptr, 125, 640, 640);
  attention_kernel<<<5, 256, 0, stream>>>(Qs, Ks, Vs, atts);
  cvt_f16_kernel<<<blocks(125 * 640 / 8), B256, 0, stream>>>(atts, attsH, 125 * 640 / 8);
  gemm_wmma<AH, BH><<<ggrid(125, 640), B256, 0, stream>>>(AH{attsH, 640}, BH{woTH, 640}, satt, nullptr, 125, 640, 640);
  blend_support<<<blocks(5 * 640 * 25), B256, 0, stream>>>(support, satt, sorig);

  // --- query fusion: concat + 1x1 conv + agent attention ---
  build_xa<<<blocks((size_t)1875 * 1280), B256, 0, stream>>>(query, sorig, XAH);
  gemm_wmma<AH, BH><<<ggrid(1875, 640), B256, 0, stream>>>(AH{XAH, 1280}, BH{convWH, 1280}, cqc, conv_b, 1875, 640, 1280);
  cvt_f16_kernel<<<blocks((size_t)1875 * 640 / 8), B256, 0, stream>>>(cqc, cqcH, 1875 * 640 / 8);
  gemm_wmma<AH, BH><<<ggrid(1875, 640), B256, 0, stream>>>(AH{cqcH, 640}, BH{wqTH, 640}, Qq, nullptr, 1875, 640, 640);
  gemm_wmma<AH, BH><<<ggrid(1875, 640), B256, 0, stream>>>(AH{cqcH, 640}, BH{wkTH, 640}, Kq, nullptr, 1875, 640, 640);
  gemm_wmma<AH, BH><<<ggrid(1875, 640), B256, 0, stream>>>(AH{cqcH, 640}, BH{wvTH, 640}, Vq, nullptr, 1875, 640, 640);
  attention_kernel<<<75, 256, 0, stream>>>(Qq, Kq, Vq, attq);
  cvt_f16_kernel<<<blocks((size_t)1875 * 640 / 8), B256, 0, stream>>>(attq, attqH, 1875 * 640 / 8);
  gemm_wmma<AH, BH><<<ggrid(1875, 640), B256, 0, stream>>>(AH{attqH, 640}, BH{woTH, 640}, qatt, nullptr, 1875, 640, 640);
  blend_query<<<blocks((size_t)75 * 640 * 25), B256, 0, stream>>>(query, qatt, qorig);

  // --- RelationNetwork (conv1 implicit-im2col WMMA GEMM dominates the FLOPs) ---
  gemm_wmma<AC1H, BH><<<ggrid(9375, 320), B256, 0, stream>>>(AC1H{ZQ}, BH{wr1, 11520}, Y1, c1_b, 9375, 320, 11520);
  bn_stats<<<320, 256, 0, stream>>>(Y1, 9375, 320, mean1, var1);
  bnpool1<<<blocks((size_t)375 * 4 * 320), B256, 0, stream>>>(Y1, mean1, var1, bn1_g, bn1_b, h1H);
  gemm_wmma<AC2H, BH><<<ggrid(1500, 64), B256, 0, stream>>>(AC2H{h1H}, BH{wr2, 2880}, Y2, c2_b, 1500, 64, 2880);
  bn_stats<<<64, 256, 0, stream>>>(Y2, 1500, 64, mean2, var2);
  bnpool2<<<blocks((size_t)375 * 64), B256, 0, stream>>>(Y2, mean2, var2, bn2_g, bn2_b, h2);
  fc_kernel<<<blocks(375), B256, 0, stream>>>(h2, fc1_w, fc1_b, fc2_w, fc2_b, score);

  // --- EMD branch ---
  pool_mean<<<blocks(5 * 640), B256, 0, stream>>>(sorig, pp, 5 * 640);
  pool_mean<<<blocks(75 * 640), B256, 0, stream>>>(qorig, qp, 75 * 640);
  compute_w1<<<blocks(9375), B256, 0, stream>>>(qorig, pp, w1b);
  compute_w2<<<blocks(9375), B256, 0, stream>>>(sorig, qp, w2b);
  normalize_q_kernel<<<1875, 128, 0, stream>>>(qorig, QNH);
  normalize_p_kernel<<<125, 128, 0, stream>>>(sorig, PNH);
  gemm_wmma<AH, BH><<<ggrid(1875, 125), B256, 0, stream>>>(AH{QNH, 640}, BH{PNH, 640}, sim2, nullptr, 1875, 125, 640);
  marginals<<<blocks(375), B256, 0, stream>>>(w1b, w2b, rm, cm);
  sinkhorn_kernel<<<375, 32, 0, stream>>>(sim2, rm, cm, score, out);

  (void)in_sizes; (void)n_in; (void)out_size; (void)ws_size;
}
